// NNConvNet_34462817583565
// MI455X (gfx1250) — matrix operand order repaired and checked by reading, not compiled, not run on hardware
//
#include <hip/hip_runtime.h>

// ---------------------------------------------------------------------------
// MI455X (gfx1250) fused NNConv: edge-MLP -> per-edge matmul -> mean-scatter
// -> relu -> edge classifier.  All heavy GEMMs via v_wmma_f32_16x16x32_bf16.
// The [E,1024] edge-conditioned weight tensor is consumed in-register and
// never written to memory.  GEMM2 is register-blocked 4(M-tiles) x 2(N-tiles)
// with the i-dimension split across waves so each W2 tile is fetched exactly
// once per workgroup.
// ---------------------------------------------------------------------------

typedef __attribute__((ext_vector_type(16))) __bf16 v16bf;
typedef __attribute__((ext_vector_type(8)))  float  v8f;

static __device__ __forceinline__ v8f wmma_bf16(v16bf a, v16bf b, v8f c) {
  // (neg_a, A, neg_b, B, c_mod, C, reuse_a, reuse_b)
  return __builtin_amdgcn_wmma_f32_16x16x32_bf16(false, a, false, b, (short)0, c,
                                                 false, false);
}

// ---------------------------------------------------------------------------
// Zero a float region (agg + deg) each launch (ws is not re-poisoned).
__global__ void k_zero_f32(float* __restrict__ p, int n) {
  int i = blockIdx.x * blockDim.x + threadIdx.x;
  if (i < n) p[i] = 0.0f;
}

// Degree of each destination node (float; exact for counts < 2^24).
__global__ void k_degree(const int* __restrict__ dst, float* __restrict__ deg, int E) {
  int i = blockIdx.x * blockDim.x + threadIdx.x;
  if (i < E) atomicAdd(deg + dst[i], 1.0f);
}

// ---------------------------------------------------------------------------
// Swizzle a row-major f32 [K x N] weight into WMMA B-operand bf16 layout.
// Tile = 32(K) x 16(N).  Within a tile: lane = half*16 + n; element s holds
// K = half*8 + s for s < 8, K = 16 + half*8 + (s-8) for s >= 8.
// Tiles ordered (ncol16-major over kt): tile_id = nc*(K/32) + kt.
__global__ void k_swizzleB(const float* __restrict__ srcw, __bf16* __restrict__ dstw,
                           int K, int N) {
  int idx = blockIdx.x * blockDim.x + threadIdx.x;
  int total = K * N;
  if (idx >= total) return;
  int ktCount = K >> 5;
  int tile = idx >> 9;          // 512 bf16 per tile
  int r    = idx & 511;
  int lane = r >> 4;            // 0..31
  int t2   = r & 15;            // element within lane
  int j    = t2 >> 1;
  int e    = t2 & 1;
  int kt   = tile % ktCount;
  int nc   = tile / ktCount;
  int half = lane >> 4;
  int nl   = lane & 15;
  int k = kt * 32 + (t2 >> 3) * 16 + half * 8 + (t2 & 7);
  int n = nc * 16 + nl;
  dstw[idx] = (__bf16)srcw[k * N + n];
}

// ---------------------------------------------------------------------------
// Main fused kernel: 64 edges per block, 8 wave32 (256 threads).
//   t  = relu(ef @ W1 + b1)                [64 x 1024]  (GEMM1, WMMA K=32)
//   w  = t @ W2 + b2 (per 32-col block i)  consumed in-register
//   msg[e,o] = sum_i x[src[e],i] * w[e, i*32+o]
//   LDS-reduce wave partials (ds_add_f32), then atomicAdd into agg[dst[e]].
__global__ __launch_bounds__(256) void k_nnconv(
    const float* __restrict__ node_feats, const float* __restrict__ edge_feats,
    const int* __restrict__ src, const int* __restrict__ dst,
    const __bf16* __restrict__ w1s, const float* __restrict__ b1,
    const __bf16* __restrict__ w2s, const float* __restrict__ b2,
    float* __restrict__ agg, int N, int E) {
  // LDS: t (A-swizzled bf16, 4 mt x 32 kt tiles) 128KB, ef (A-swizzled) 4KB,
  //      x (row-major f32, stride 33) 8.25KB, msg reduction buffer 8.25KB.
  __shared__ __align__(32) __bf16 t_s[128 * 512];
  __shared__ __align__(32) __bf16 ef_s[4 * 512];
  __shared__ float x_s[64 * 33];
  __shared__ float msg_s[64 * 33];
  __shared__ int  src_s[64];
  __shared__ int  dst_s[64];

  const int tid  = threadIdx.x;
  const int lane = tid & 31;
  const int w    = tid >> 5;
  const int hc   = lane >> 4;   // C-layout half
  const int ln   = lane & 15;   // C-layout column / B-layout column
  const int e0   = blockIdx.x * 64;

  // ---- Phase A1: gather src/dst + edge_feats (A-swizzled bf16) ------------
  for (int idx = tid; idx < 64; idx += 256) {
    int ge = e0 + idx;
    src_s[idx] = (ge < E) ? src[ge] : 0;
    dst_s[idx] = (ge < E) ? dst[ge] : 0;
  }
  for (int idx = tid; idx < 64 * 32; idx += 256) {
    int e_l = idx >> 5, f = idx & 31, ge = e0 + e_l;
    float v = (ge < E) ? edge_feats[ge * 32 + f] : 0.0f;
    // A-layout: element (m,k): lane = m + 16*((k%16)/8), slot = (k/16)*4+(k%8)/2, pair k%2
    int mt = e_l >> 4, m = e_l & 15;
    int half = (f >> 3) & 1, jv = (f >> 4) * 4 + ((f & 7) >> 1), ev = f & 1;
    ef_s[mt * 512 + (m + 16 * half) * 16 + jv * 2 + ev] = (__bf16)v;
  }
  __syncthreads();

  // ---- Phase A2: gather x = node_feats[src]; zero the msg buffer ----------
  for (int idx = tid; idx < 64 * 32; idx += 256) {
    int e_l = idx >> 5, f = idx & 31, ge = e0 + e_l;
    x_s[e_l * 33 + f] = (ge < E) ? node_feats[src_s[e_l] * 32 + f] : 0.0f;
  }
  for (int idx = tid; idx < 64 * 33; idx += 256) msg_s[idx] = 0.0f;

  // ---- GEMM1: t = relu(ef @ W1 + b1), stored A-swizzled bf16 in LDS -------
  {
    const int mt1 = w & 3;     // which 16-edge group this wave owns
    const int p   = w >> 2;    // n-tile parity
    v16bf a_ef = *(const v16bf*)(ef_s + mt1 * 512 + lane * 16);
    for (int j2 = p; j2 < 64; j2 += 2) {         // 16-col tiles of t
      v16bf b = *(const v16bf*)(w1s + j2 * 512 + lane * 16);
      v8f c = {};
      c = wmma_bf16(a_ef, b, c);
      float b1n = b1[j2 * 16 + ln];
      // store: t column k = j2*16+ln -> A-tile (mt, j2/2), k_local = (j2&1)*16+ln
      int kt2   = j2 >> 1;
      int halfA = ln >> 3;
      int slot  = ((j2 & 1) * 4 + ((ln & 7) >> 1)) * 2 + (ln & 1);
      __bf16* tb = t_s + (mt1 * 32 + kt2) * 512;
#pragma unroll
      for (int q = 0; q < 8; ++q) {
        float v = c[q] + b1n;
        v = v > 0.0f ? v : 0.0f;
        int lane_t = (q + 8 * hc) + 16 * halfA;
        tb[lane_t * 16 + slot] = (__bf16)v;
      }
    }
  }
  __syncthreads();

  // ---- GEMM2 + per-edge contraction ---------------------------------------
  // Wave w owns i in {w, w+8, w+16, w+24}: each W2 tile is read exactly once
  // per workgroup.  Register block: 4 M-tiles (all 64 edges) x 2 N-tiles.
  float msgacc[4][2][8];
#pragma unroll
  for (int mt = 0; mt < 4; ++mt)
#pragma unroll
    for (int nt = 0; nt < 2; ++nt)
#pragma unroll
      for (int q = 0; q < 8; ++q) msgacc[mt][nt][q] = 0.0f;

  for (int i = w; i < 32; i += 8) {
    v8f C[4][2];
#pragma unroll
    for (int mt = 0; mt < 4; ++mt) {
      C[mt][0] = (v8f){};
      C[mt][1] = (v8f){};
    }
    const __bf16* bb0p = w2s + ((i * 2 + 0) * 32) * 512 + lane * 16;
    const __bf16* bb1p = w2s + ((i * 2 + 1) * 32) * 512 + lane * 16;
    const __bf16* ab   = t_s + lane * 16;
#pragma unroll 2
    for (int kt = 0; kt < 32; ++kt) {
      v16bf b0 = *(const v16bf*)(bb0p + kt * 512);
      v16bf b1v = *(const v16bf*)(bb1p + kt * 512);
#pragma unroll
      for (int mt = 0; mt < 4; ++mt) {
        v16bf a = *(const v16bf*)(ab + (mt * 32 + kt) * 512);
        C[mt][0] = wmma_bf16(a, b0, C[mt][0]);
        C[mt][1] = wmma_bf16(a, b1v, C[mt][1]);
      }
    }
    float bb0 = b2[i * 32 + ln];
    float bb1 = b2[i * 32 + 16 + ln];
#pragma unroll
    for (int mt = 0; mt < 4; ++mt)
#pragma unroll
      for (int q = 0; q < 8; ++q) {
        float xq = x_s[(mt * 16 + q + 8 * hc) * 33 + i];
        msgacc[mt][0][q] = fmaf(xq, C[mt][0][q] + bb0, msgacc[mt][0][q]);
        msgacc[mt][1][q] = fmaf(xq, C[mt][1][q] + bb1, msgacc[mt][1][q]);
      }
  }
  // Reduce wave partials in LDS (ds_add_f32).
#pragma unroll
  for (int mt = 0; mt < 4; ++mt)
#pragma unroll
    for (int q = 0; q < 8; ++q) {
      int e_row = mt * 16 + q + 8 * hc;
      atomicAdd(&msg_s[e_row * 33 + ln],      msgacc[mt][0][q]);
      atomicAdd(&msg_s[e_row * 33 + 16 + ln], msgacc[mt][1][q]);
    }
  __syncthreads();

  // Minimal global scatter: one atomic per (edge, out-feature).
  for (int idx = tid; idx < 64 * 32; idx += 256) {
    int e_l = idx >> 5, o = idx & 31, ge = e0 + e_l;
    if (ge < E) atomicAdd(agg + dst_s[e_l] * 32 + o, msg_s[e_l * 33 + o]);
  }
}

// ---------------------------------------------------------------------------
// h = relu(agg / max(deg,1) + conv_bias)
__global__ void k_node_act(const float* __restrict__ agg, const float* __restrict__ deg,
                           const float* __restrict__ bias, float* __restrict__ h, int n) {
  int i = blockIdx.x * blockDim.x + threadIdx.x;
  if (i >= n) return;
  float d = deg[i >> 5];
  float v = agg[i] / fmaxf(d, 1.0f) + bias[i & 31];
  h[i] = fmaxf(v, 0.0f);
}

// ---------------------------------------------------------------------------
// Edge classifier: concat(h[s],h[d],ef) [E x 96] @ Wc1 (WMMA, K=96) -> relu
// -> @ Wc2 [32x3] (VALU f32).  128 edges / block, 16 edges per wave.
__global__ __launch_bounds__(256) void k_classify(
    const float* __restrict__ h, const float* __restrict__ edge_feats,
    const int* __restrict__ src, const int* __restrict__ dst,
    const int* __restrict__ eidx, const __bf16* __restrict__ wc1s,
    const float* __restrict__ bc1, const float* __restrict__ Wc2,
    const float* __restrict__ bc2, float* __restrict__ out, int E) {
  __shared__ __align__(32) __bf16 aw[8 * 1536];  // 3 A-tiles (K=96) per wave
  __shared__ float h1w[8 * 16 * 33];
  __shared__ int sw[8 * 16], dw[8 * 16], ew[8 * 16];

  const int tid = threadIdx.x;
  const int lane = tid & 31;
  const int w = tid >> 5;
  const int hc = lane >> 4, ln = lane & 15;
  const int e0w = blockIdx.x * 128 + w * 16;

  if (lane < 16) {
    int ge = e0w + lane;
    int ok = ge < E;
    int ei = ok ? eidx[ge] : 0;
    ew[w * 16 + lane] = ei;
    sw[w * 16 + lane] = ok ? src[ei] : 0;
    dw[w * 16 + lane] = ok ? dst[ei] : 0;
  }
  __syncthreads();

  // Build A [16 x 96] in A-swizzled bf16 layout.
  for (int idx = lane; idx < 16 * 96; idx += 32) {
    int e_l = idx & 15, f = idx >> 4, ge = e0w + e_l;
    float v = 0.0f;
    if (ge < E) {
      if (f < 32)      v = h[sw[w * 16 + e_l] * 32 + f];
      else if (f < 64) v = h[dw[w * 16 + e_l] * 32 + (f - 32)];
      else             v = edge_feats[ew[w * 16 + e_l] * 32 + (f - 64)];
    }
    int kt = f >> 5, kl = f & 31;
    int half = (kl >> 3) & 1, jv = (kl >> 4) * 4 + ((kl & 7) >> 1), ev = kl & 1;
    aw[w * 1536 + kt * 512 + (e_l + 16 * half) * 16 + jv * 2 + ev] = (__bf16)v;
  }
  __syncthreads();

  v8f c0 = {}, c1 = {};
#pragma unroll
  for (int kt = 0; kt < 3; ++kt) {
    v16bf a  = *(const v16bf*)(aw + w * 1536 + kt * 512 + lane * 16);
    v16bf b0 = *(const v16bf*)(wc1s + (0 * 3 + kt) * 512 + lane * 16);
    v16bf b1 = *(const v16bf*)(wc1s + (1 * 3 + kt) * 512 + lane * 16);
    c0 = wmma_bf16(a, b0, c0);
    c1 = wmma_bf16(a, b1, c1);
  }
  float bcl0 = bc1[ln], bcl1 = bc1[16 + ln];
#pragma unroll
  for (int q = 0; q < 8; ++q) {
    int m = q + 8 * hc;
    float v0 = fmaxf(c0[q] + bcl0, 0.0f);
    float v1 = fmaxf(c1[q] + bcl1, 0.0f);
    h1w[(w * 16 + m) * 33 + ln]      = v0;
    h1w[(w * 16 + m) * 33 + 16 + ln] = v1;
  }
  __syncthreads();

  // Layer 2: [16 x 32] @ [32 x 3] in f32.
  for (int t = lane; t < 48; t += 32) {
    int e_l = t & 15, cc = t >> 4, ge = e0w + e_l;
    if (ge >= E) continue;
    const float* hr = h1w + (w * 16 + e_l) * 33;
    float acc = bc2[cc];
#pragma unroll
    for (int n = 0; n < 32; ++n) acc = fmaf(hr[n], Wc2[n * 3 + cc], acc);
    out[ge * 3 + cc] = acc;
  }
}

// ---------------------------------------------------------------------------
extern "C" void kernel_launch(void* const* d_in, const int* in_sizes, int n_in,
                              void* d_out, int out_size, void* d_ws, size_t ws_size,
                              hipStream_t stream) {
  const float* node_feats = (const float*)d_in[0];
  const float* edge_feats = (const float*)d_in[1];
  const int*   src        = (const int*)d_in[2];
  const int*   dst        = (const int*)d_in[3];
  const int*   eidx       = (const int*)d_in[4];
  const float* W1         = (const float*)d_in[5];
  const float* b1         = (const float*)d_in[6];
  const float* W2         = (const float*)d_in[7];
  const float* b2         = (const float*)d_in[8];
  const float* conv_bias  = (const float*)d_in[9];
  const float* Wc1        = (const float*)d_in[10];
  const float* bc1        = (const float*)d_in[11];
  const float* Wc2        = (const float*)d_in[12];
  const float* bc2        = (const float*)d_in[13];
  float* out = (float*)d_out;

  const int N = in_sizes[0] / 32;
  const int E = in_sizes[2];

  auto al = [](size_t x) { return (x + 255) & ~(size_t)255; };
  char* wsb = (char*)d_ws;
  size_t off_agg  = 0;
  size_t off_deg  = off_agg + (size_t)N * 32 * sizeof(float);   // contiguous w/ agg
  size_t off_h    = al(off_deg + (size_t)N * sizeof(float));
  size_t off_w1s  = al(off_h + (size_t)N * 32 * sizeof(float));
  size_t off_w2s  = al(off_w1s + (size_t)32 * 1024 * sizeof(__bf16));
  size_t off_wc1s = al(off_w2s + (size_t)1024 * 1024 * sizeof(__bf16));
  float*  agg  = (float*)(wsb + off_agg);
  float*  deg  = (float*)(wsb + off_deg);
  float*  hbuf = (float*)(wsb + off_h);
  __bf16* w1s  = (__bf16*)(wsb + off_w1s);
  __bf16* w2s  = (__bf16*)(wsb + off_w2s);
  __bf16* wc1s = (__bf16*)(wsb + off_wc1s);

  // 1) zero agg + deg (contiguous N*33 floats)
  {
    int n = N * 33;
    k_zero_f32<<<(n + 255) / 256, 256, 0, stream>>>(agg, n);
  }
  // 2) degrees
  k_degree<<<(E + 255) / 256, 256, 0, stream>>>(dst, deg, E);
  // 3) weight swizzles (bf16 B-operand layout)
  k_swizzleB<<<(32 * 1024 + 255) / 256, 256, 0, stream>>>(W1, w1s, 32, 1024);
  k_swizzleB<<<(1024 * 1024 + 255) / 256, 256, 0, stream>>>(W2, w2s, 1024, 1024);
  k_swizzleB<<<(96 * 32 + 255) / 256, 256, 0, stream>>>(Wc1, wc1s, 96, 32);
  // 4) fused NNConv (edge MLP + per-edge matmul + scatter)
  k_nnconv<<<(E + 63) / 64, 256, 0, stream>>>(node_feats, edge_feats, src, dst,
                                              w1s, b1, w2s, b2, agg, N, E);
  // 5) node activation
  {
    int n = N * 32;
    k_node_act<<<(n + 255) / 256, 256, 0, stream>>>(agg, deg, conv_bias, hbuf, n);
  }
  // 6) edge classifier
  k_classify<<<(E + 127) / 128, 256, 0, stream>>>(hbuf, edge_feats, src, dst, eidx,
                                                  wc1s, bc1, Wc2, bc2, out, E);
}
